// SelfAttention_85023172592578
// MI455X (gfx1250) — matrix operand order repaired and checked by reading, not compiled, not run on hardware
//
#include <hip/hip_runtime.h>
#include <hip/hip_bf16.h>
#include <math.h>

// ---------------------------------------------------------------------------
// Problem constants (from the reference): B=4, S=2048, D=512, H=8, HD=64
// ---------------------------------------------------------------------------
constexpr int kB  = 4;
constexpr int kS  = 2048;
constexpr int kD  = 512;
constexpr int kH  = 8;
constexpr int kHD = 64;
constexpr float kScale = 0.044194173824159216f;  // 1/sqrt(512)  (reference scales by sqrt(D))

typedef __attribute__((ext_vector_type(16))) _Float16 v16h;
typedef __attribute__((ext_vector_type(8)))  _Float16 v8h;
typedef __attribute__((ext_vector_type(4)))  _Float16 v4h;
typedef __attribute__((ext_vector_type(8)))  float    v8f;
typedef __attribute__((ext_vector_type(4)))  float    v4f;
typedef __attribute__((ext_vector_type(4)))  int      v4i;

// gfx1250 async global->LDS copy (ASYNCcnt-tracked); guarded so we always compile.
#if defined(__AMDGCN__) && __has_builtin(__builtin_amdgcn_global_load_async_to_lds_b128)
#define USE_ASYNC_LDS 1
#else
#define USE_ASYNC_LDS 0
#endif

__device__ __forceinline__ void cp_async16(const _Float16* g, _Float16* l) {
#if USE_ASYNC_LDS
  // prototype (probe-verified): (v4i as1*, v4i as3*, imm offset, imm cpol)
  __builtin_amdgcn_global_load_async_to_lds_b128(
      (__attribute__((address_space(1))) v4i*)g,
      (__attribute__((address_space(3))) v4i*)l, /*offset=*/0, /*cpol=*/0);
#else
  *(v8h*)l = *(const v8h*)g;   // sync fallback: global load + ds_store
#endif
}

__device__ __forceinline__ void wait_async() {
#if USE_ASYNC_LDS
#if __has_builtin(__builtin_amdgcn_s_wait_asynccnt)
  __builtin_amdgcn_s_wait_asynccnt(0);
#else
  asm volatile("s_wait_asynccnt 0x0" ::: "memory");
#endif
#endif
}

// D = A(16x32 f16) x B(32x16 f16) + C(16x16 f32)
__device__ __forceinline__ v8f wmma16(v16h a, v16h b, v8f c) {
  return __builtin_amdgcn_wmma_f32_16x16x32_f16(false, a, false, b, (short)0, c,
                                                false, false);
}

// ---- A-operand loaders (ISA 7.12.2, 16-bit A 16x32):
//   lane<16 : halves 0..7 -> K=kbase+0..7,   halves 8..15 -> K=kbase+16..23
//   lane>=16: halves 0..7 -> K=kbase+8..15,  halves 8..15 -> K=kbase+24..31
__device__ __forceinline__ v16h load_a_f16(const _Float16* rowptr, int hi, int kbase) {
  const _Float16* p = rowptr + kbase + hi * 8;
  v8h a = *(const v8h*)p;
  v8h b = *(const v8h*)(p + 16);
  v16h r;
#pragma unroll
  for (int i = 0; i < 8; ++i) { r[i] = a[i]; r[i + 8] = b[i]; }
  return r;
}
__device__ __forceinline__ v16h load_a_f32(const float* rowptr, int hi, int kbase) {
  const float* p = rowptr + kbase + hi * 8;
  v16h r;
#pragma unroll
  for (int i = 0; i < 8; ++i) {
    r[i]     = (_Float16)p[i];
    r[i + 8] = (_Float16)p[i + 16];
  }
  return r;
}

// ---- B-operand loaders (same K order as the A loaders):
//   lane<16 : halves 0..7 -> K=kbase+0..7, 8..15 -> K=kbase+16..23 ; lane>=16: K=kbase+8..15, 24..31
__device__ __forceinline__ v16h load_b_f16(const _Float16* rowptr, int hi, int kbase) {
  const _Float16* p = rowptr + kbase + 8 * hi;
  v8h a = *(const v8h*)p;
  v8h b = *(const v8h*)(p + 16);
  v16h r;
#pragma unroll
  for (int i = 0; i < 8; ++i) { r[i] = a[i]; r[i + 8] = b[i]; }
  return r;
}
__device__ __forceinline__ v16h load_b_f32(const float* rowptr, int hi, int kbase) {
  const float* p = rowptr + kbase + 8 * hi;
  v16h r;
#pragma unroll
  for (int i = 0; i < 8; ++i) { r[i] = (_Float16)p[i]; r[i + 8] = (_Float16)p[i + 16]; }
  return r;
}

// ---------------------------------------------------------------------------
// Kernel 0: one-time f32 -> f16 weight conversion (Wo, 512x512)
// ---------------------------------------------------------------------------
__global__ void cvt_f16_kernel(const float* __restrict__ src, _Float16* __restrict__ dst) {
  const int i = (blockIdx.x * blockDim.x + threadIdx.x) * 4;
  const v4f v = *(const v4f*)(src + i);
  v4h o;
#pragma unroll
  for (int j = 0; j < 4; ++j) o[j] = (_Float16)v[j];
  *(v4h*)(dst + i) = o;
}

// ---------------------------------------------------------------------------
// Kernel 1: per-head linear projection. out = (x_head @ W^T + bias) * scale
//   transposed==0: out f16 [B,H,S,HD] (token-major)   -- Q, K
//   transposed==1: out f16 [B,H,HD,S] (feature-major) -- V
// grid: (B*S/16, H), block 128 (4 waves, one 16-col tile each)
// ---------------------------------------------------------------------------
__global__ void qkv_proj_kernel(const float* __restrict__ x, const float* __restrict__ W,
                                const float* __restrict__ bias, _Float16* __restrict__ out,
                                int transposed, float scale) {
  const int tid  = threadIdx.x;
  const int wave = tid >> 5;
  const int lane = tid & 31;
  const int hi   = lane >> 4;
  const int ln   = lane & 15;
  const int tbase = blockIdx.x * 16;
  const int h     = blockIdx.y;
  const int cbase = wave * 16;

  const float* arow = x + (size_t)(tbase + ln) * kD + h * kHD;
  const float* brow = W + (size_t)(cbase + ln) * kHD;

  v8f acc;
  const float bv = bias[cbase + ln];
#pragma unroll
  for (int i = 0; i < 8; ++i) acc[i] = bv;

#pragma unroll
  for (int ks = 0; ks < 2; ++ks) {
    v16h a = load_a_f32(arow, hi, ks * 32);
    v16h b = load_b_f32(brow, hi, ks * 32);
    acc = wmma16(a, b, acc);
  }

  const int b_idx  = tbase / kS;
  const int s_in_b = tbase - b_idx * kS;
  if (!transposed) {
    _Float16* o = out + ((size_t)(b_idx * kH + h) * kS + s_in_b) * kHD + cbase + ln;
#pragma unroll
    for (int v = 0; v < 8; ++v)
      o[(size_t)(v + 8 * hi) * kHD] = (_Float16)(acc[v] * scale);
  } else {
    _Float16* o = out + ((size_t)(b_idx * kH + h) * kHD + cbase + ln) * kS + s_in_b + 8 * hi;
    v8h pk;
#pragma unroll
    for (int v = 0; v < 8; ++v) pk[v] = (_Float16)(acc[v] * scale);
    *(v8h*)o = pk;
  }
}

// ---------------------------------------------------------------------------
// Kernel 2: flash attention. One 16-query tile per wave; the block's 4 waves
// share double-buffered LDS K/V chunks (32 keys) filled with async LDS DMA so
// the next chunk's copy overlaps this chunk's 8 WMMAs + softmax.
//   S^T = K @ Q^T, so softmax stats are per-lane (+1 shfl_xor(16)) and the
//   C-layout of S^T is byte-identical to the A-operand layout of P.
// grid: (S/64, H, B), block 128
// ---------------------------------------------------------------------------
__global__ void attn_kernel(const _Float16* __restrict__ Qh, const _Float16* __restrict__ Kh,
                            const _Float16* __restrict__ Vt, _Float16* __restrict__ attn_out) {
  __shared__ __align__(16) _Float16 Ksh[2][32][kHD];   // [buf][key][d]      4KB each
  __shared__ __align__(16) _Float16 Vsh[2][kHD][32];   // [buf][d][key]      4KB each

  const int tid  = threadIdx.x;
  const int wave = tid >> 5;
  const int lane = tid & 31;
  const int hi   = lane >> 4;
  const int ln   = lane & 15;
  const int qtile = blockIdx.x * 4 + wave;
  const int h = blockIdx.y;
  const int b = blockIdx.z;
  const int qbase = qtile * 16;
  const size_t bh = (size_t)(b * kH + h);

  // stage one 32-key chunk (K: 32x64 f16, V^T: 64x32 f16) -> 4 x b128 per thread
  auto stage = [&](int jb, int buf) {
#pragma unroll
    for (int i = 0; i < 2; ++i) {
      const int sidx = tid + i * 128;            // 0..255 16B segments
      const int krow = sidx >> 3, kseg = sidx & 7;
      cp_async16(Kh + (bh * kS + jb + krow) * kHD + kseg * 8, &Ksh[buf][krow][kseg * 8]);
      const int vrow = sidx >> 2, vseg = sidx & 3;
      cp_async16(Vt + (bh * kHD + vrow) * kS + jb + vseg * 8, &Vsh[buf][vrow][vseg * 8]);
    }
  };

  // Q fragment (B operand of score WMMA): lane -> query qbase+ln, contiguous d
  const _Float16* qrow = Qh + (bh * kS + qbase + ln) * kHD;
  const v16h qf0 = load_b_f16(qrow, hi, 0);
  const v16h qf1 = load_b_f16(qrow, hi, 32);

  float m_run = -INFINITY;
  float l_run = 0.f;
  v8f acc[4];
#pragma unroll
  for (int t = 0; t < 4; ++t)
#pragma unroll
    for (int i = 0; i < 8; ++i) acc[t][i] = 0.f;

  v8f zero;
#pragma unroll
  for (int i = 0; i < 8; ++i) zero[i] = 0.f;

  stage(0, 0);
  int buf = 0;
  for (int jb = 0; jb < kS; jb += 32) {
    wait_async();        // our async writes into Ksh/Vsh[buf] have landed
    __syncthreads();     // everyone's writes visible; prior reads of buf^1 done
    if (jb + 32 < kS) stage(jb + 32, buf ^ 1);

    // ---- scores (transposed): ST[kj, qi], two 16-key tiles (from LDS)
    const _Float16* krow0 = &Ksh[buf][ln][0];
    const _Float16* krow1 = &Ksh[buf][16 + ln][0];
    v8f st0 = wmma16(load_a_f16(krow0, hi, 0),  qf0, zero);
    st0     = wmma16(load_a_f16(krow0, hi, 32), qf1, st0);
    v8f st1 = wmma16(load_a_f16(krow1, hi, 0),  qf0, zero);
    st1     = wmma16(load_a_f16(krow1, hi, 32), qf1, st1);

    // ---- online softmax over this 32-key chunk (scale pre-folded into Qh)
    float mc = st0[0];
#pragma unroll
    for (int i = 1; i < 8; ++i) mc = fmaxf(mc, st0[i]);
#pragma unroll
    for (int i = 0; i < 8; ++i) mc = fmaxf(mc, st1[i]);
    mc = fmaxf(mc, __shfl_xor(mc, 16, 32));
    const float m_new = fmaxf(m_run, mc);
    const float alpha = __expf(m_run - m_new);   // first iter: exp(-inf)=0

    v16h pa;                                     // P in A-operand layout (no shuffles)
    float r = 0.f;
#pragma unroll
    for (int i = 0; i < 8; ++i) {
      const float p0 = __expf(st0[i] - m_new);
      const float p1 = __expf(st1[i] - m_new);
      r += p0 + p1;
      pa[i]     = (_Float16)p0;
      pa[i + 8] = (_Float16)p1;
    }
    r += __shfl_xor(r, 16, 32);
    l_run = l_run * alpha + r;
    m_run = m_new;

    // ---- rescale O accumulators by per-query alpha (alpha lives in lane qi)
#pragma unroll
    for (int v = 0; v < 8; ++v) {
      const float av = __shfl(alpha, v + 8 * hi, 32);
#pragma unroll
      for (int t = 0; t < 4; ++t) acc[t][v] *= av;
    }

    // ---- O += P @ V  (V^T rows in LDS give contiguous B-operand loads)
#pragma unroll
    for (int t = 0; t < 4; ++t) {
      v16h vf = load_b_f16(&Vsh[buf][t * 16 + ln][0], hi, 0);
      acc[t] = wmma16(pa, vf, acc[t]);
    }
    buf ^= 1;
  }

  // ---- normalize by l and store f16 to [B*S, D] token-major
#pragma unroll
  for (int v = 0; v < 8; ++v) {
    const float lv  = __shfl(l_run, v + 8 * hi, 32);
    const float inv = 1.f / lv;
    const int qi = v + 8 * hi;
    const size_t tok = (size_t)b * kS + qbase + qi;
#pragma unroll
    for (int t = 0; t < 4; ++t)
      attn_out[tok * kD + h * kHD + t * 16 + ln] = (_Float16)(acc[t][v] * inv);
  }
}

// ---------------------------------------------------------------------------
// Kernel 3a: output projection. lin = attn_out @ Wo16^T + bo   (f32 result)
// grid: (B*S/16, D/64), block 128, K-loop 512/32 = 16 WMMAs
// ---------------------------------------------------------------------------
__global__ void out_proj_kernel(const _Float16* __restrict__ ain, const _Float16* __restrict__ Wo16,
                                const float* __restrict__ bo, float* __restrict__ lin) {
  const int tid  = threadIdx.x;
  const int wave = tid >> 5;
  const int lane = tid & 31;
  const int hi   = lane >> 4;
  const int ln   = lane & 15;
  const int tbase = blockIdx.x * 16;
  const int cbase = (blockIdx.y * 4 + wave) * 16;

  const _Float16* arow = ain  + (size_t)(tbase + ln) * kD;
  const _Float16* brow = Wo16 + (size_t)(cbase + ln) * kD;

  v8f acc;
  const float bv = bo[cbase + ln];
#pragma unroll
  for (int i = 0; i < 8; ++i) acc[i] = bv;

#pragma unroll
  for (int ks = 0; ks < 16; ++ks) {
    v16h a = load_a_f16(arow, hi, ks * 32);
    v16h b = load_b_f16(brow, hi, ks * 32);
    acc = wmma16(a, b, acc);
  }

  float* o = lin + (size_t)tbase * kD + cbase + ln;
#pragma unroll
  for (int v = 0; v < 8; ++v) o[(size_t)(v + 8 * hi) * kD] = acc[v];
}

// ---------------------------------------------------------------------------
// Kernel 3b: out = q + LayerNorm(lin) * gamma + beta   (one token per block)
// ---------------------------------------------------------------------------
__global__ void ln_res_kernel(const float* __restrict__ lin, const float* __restrict__ qin,
                              const float* __restrict__ gamma, const float* __restrict__ beta,
                              float* __restrict__ out) {
  const int t   = blockIdx.x;
  const int tid = threadIdx.x;  // 256 threads, 2 elems each
  const size_t base = (size_t)t * kD;
  const float x0 = lin[base + tid];
  const float x1 = lin[base + tid + 256];

  float s  = x0 + x1;
  float ss = x0 * x0 + x1 * x1;
#pragma unroll
  for (int m = 16; m >= 1; m >>= 1) {
    s  += __shfl_xor(s,  m, 32);
    ss += __shfl_xor(ss, m, 32);
  }
  __shared__ float sm1[8], sm2[8];
  const int wave = tid >> 5, lane = tid & 31;
  if (lane == 0) { sm1[wave] = s; sm2[wave] = ss; }
  __syncthreads();
  float S1 = 0.f, S2 = 0.f;
#pragma unroll
  for (int i = 0; i < 8; ++i) { S1 += sm1[i]; S2 += sm2[i]; }

  const float mu   = S1 * (1.0f / kD);
  const float var  = S2 * (1.0f / kD) - mu * mu;
  const float rstd = rsqrtf(var + 1e-5f);

  out[base + tid]       = qin[base + tid]       + (x0 - mu) * rstd * gamma[tid]       + beta[tid];
  out[base + tid + 256] = qin[base + tid + 256] + (x1 - mu) * rstd * gamma[tid + 256] + beta[tid + 256];
}

// ---------------------------------------------------------------------------
// Launch. Workspace layout:
//   [0,     16 MB) Qh f16 + Kh f16        -- later reused by lin f32
//   [16 MB, 24 MB) Vt f16
//   [24 MB, 32 MB) attn_out f16
//   [32 MB, 32.5M) Wo16 f16
// ---------------------------------------------------------------------------
extern "C" void kernel_launch(void* const* d_in, const int* in_sizes, int n_in,
                              void* d_out, int out_size, void* d_ws, size_t ws_size,
                              hipStream_t stream) {
  (void)in_sizes; (void)n_in; (void)out_size; (void)ws_size;
  const float* q     = (const float*)d_in[0];
  const float* k     = (const float*)d_in[1];
  const float* v     = (const float*)d_in[2];
  const float* Wq    = (const float*)d_in[3];
  const float* bq    = (const float*)d_in[4];
  const float* Wk    = (const float*)d_in[5];
  const float* bk    = (const float*)d_in[6];
  const float* Wv    = (const float*)d_in[7];
  const float* bv    = (const float*)d_in[8];
  const float* Wo    = (const float*)d_in[9];
  const float* bo    = (const float*)d_in[10];
  const float* gamma = (const float*)d_in[11];
  const float* beta  = (const float*)d_in[12];
  float* outp = (float*)d_out;

  const size_t nBH = (size_t)kB * kH * kS * kHD;   // 4,194,304 elems
  char* ws = (char*)d_ws;
  _Float16* Qh   = (_Float16*)ws;
  _Float16* Kh   = Qh + nBH;
  _Float16* Vt   = Kh + nBH;
  _Float16* aout = Vt + nBH;
  _Float16* Wo16 = aout + nBH;
  float*    lin  = (float*)ws;                     // overlaps Qh/Kh (dead by then)

  const dim3 blk(128);
  cvt_f16_kernel<<<dim3(kD * kD / 1024), dim3(256), 0, stream>>>(Wo, Wo16);
  qkv_proj_kernel<<<dim3(kB * kS / 16, kH), blk, 0, stream>>>(q, Wq, bq, Qh, 0, kScale);
  qkv_proj_kernel<<<dim3(kB * kS / 16, kH), blk, 0, stream>>>(k, Wk, bk, Kh, 0, 1.0f);
  qkv_proj_kernel<<<dim3(kB * kS / 16, kH), blk, 0, stream>>>(v, Wv, bv, Vt, 1, 1.0f);
  attn_kernel<<<dim3(kS / 64, kH, kB), blk, 0, stream>>>(Qh, Kh, Vt, aout);
  out_proj_kernel<<<dim3(kB * kS / 16, kD / 64), blk, 0, stream>>>(aout, Wo16, bo, lin);
  ln_res_kernel<<<dim3(kB * kS), dim3(256), 0, stream>>>(lin, q, gamma, beta, outp);
}
